// MessageLayer_55241869361626
// MI455X (gfx1250) — matrix operand order repaired and checked by reading, not compiled, not run on hardware
//
#include <hip/hip_runtime.h>

// ---------------------------------------------------------------------------
// GATv2 (H=4, C=128) + LayerNorm + FFN for MI455X (gfx1250, wave32, WMMA).
// - GEMMs use v_wmma_f32_16x16x32_f16 (f16 operands, fp32 accumulation).
// - Weights pre-swizzled into WMMA B-fragment-major order: one aligned 32B
//   load (2x global_load_b128) per fragment per lane; fragments for a whole
//   K-sweep are batch-loaded ahead of the WMMA chain so loads overlap math.
// - A tiles staged in LDS in A-fragment order (2x ds_load_b128 per frag).
// - Full-tile fast path for D stores (uniform branch) keeps per-row
//   exec-mask ladders off the hot path.
// - xl/xr projections stored f16 (102 MB total) so both fit in the 192 MB L2,
//   making the random per-edge gathers L2-resident (the actual bottleneck).
// ---------------------------------------------------------------------------

typedef __attribute__((ext_vector_type(16))) _Float16 v16h;
typedef __attribute__((ext_vector_type(8)))  float    v8f;

#define NEG_SLOPE 0.2f

__device__ __forceinline__ float lrelu(float v) { return v > 0.f ? v : NEG_SLOPE * v; }

// Order-preserving float -> uint mapping so segment-max can use u32 atomicMax.
__device__ __forceinline__ unsigned fkey(float x) {
    unsigned b = __float_as_uint(x);
    return (b & 0x80000000u) ? ~b : (b | 0x80000000u);
}
__device__ __forceinline__ float funkey(unsigned k) {
    unsigned b = (k & 0x80000000u) ? (k & 0x7fffffffu) : ~k;
    return __uint_as_float(b);
}
#define FKEY_NEG_INF 0x007FFFFFu  // fkey(-inf)

// --- WMMA fragment index helpers (ISA 7.12.2 16-bit layouts, wave32) --------
// B 32x16 f16 fragment: lane = (((kk>>3)&1)<<4) + (n&15), half index = ((kk>>4)<<3) + (kk&7).
// Fragment-major storage: pos = ((nt*kchunks + kc)*32 + lane)*16 + idx
__device__ __forceinline__ size_t bswz_pos(int k, int n, int kchunks) {
    int nt = n >> 4, n15 = n & 15;
    int kc = k >> 5, kk = k & 31;
    int lane = (((kk >> 3) & 1) << 4) + n15;
    int idx  = ((kk >> 4) << 3) + (kk & 7);
    return (((size_t)nt * kchunks + kc) * 32 + lane) * 16 + idx;
}
// A 16x32 f16 fragment: lane = (((kk>>3)&1)<<4) + m, idx = ((kk>>4)<<3) + (kk&7)
__device__ __forceinline__ int aswz_pos(int m, int k) {
    int kc = k >> 5, kk = k & 31;
    int lane = (((kk >> 3) & 1) << 4) + m;
    int idx  = ((kk >> 4) << 3) + (kk & 7);
    return ((kc << 5) + lane) * 16 + idx;
}

__device__ __forceinline__ v16h load_b_swz(const _Float16* Ws, int kchunks, int nt, int kc, int lane) {
    return ((const v16h*)Ws)[((size_t)nt * kchunks + kc) * 32 + lane];
}
__device__ __forceinline__ v8f wmma_f16(v16h a, v16h b, v8f c) {
    return __builtin_amdgcn_wmma_f32_16x16x32_f16(false, a, false, b, (short)0, c, false, false);
}

// --- 1. weight conversion f32 -> f16, swizzled to B-fragment order ---------
__global__ void prep_weights(const float* Wl, const float* Wr, const float* W1, const float* W2,
                             _Float16* Wl16, _Float16* Wr16, _Float16* W1h, _Float16* W2h) {
    int t = blockIdx.x * blockDim.x + threadIdx.x;
    if (t < 128 * 512) {  // K=128, N=512, kchunks=4
        int k = t >> 9, n = t & 511;
        size_t p = bswz_pos(k, n, 4);
        Wl16[p] = (_Float16)Wl[t];
        Wr16[p] = (_Float16)Wr[t];
    }
    if (t < 128 * 256) {  // K=128, N=256, kchunks=4
        int k = t >> 8, n = t & 255;
        W1h[bswz_pos(k, n, 4)] = (_Float16)W1[t];
    }
    if (t < 256 * 128) {  // K=256, N=128, kchunks=8
        int k = t >> 7, n = t & 127;
        W2h[bswz_pos(k, n, 8)] = (_Float16)W2[t];
    }
}

// --- 2. init segment-max keys / denominators / accumulators ----------------
__global__ void init_ws(unsigned* amaxu, float* denom, float* acc, int N) {
    int t = blockIdx.x * blockDim.x + threadIdx.x;
    if (t < N * 4) { amaxu[t] = FKEY_NEG_INF; denom[t] = 0.f; }
    if (t < N * 128) acc[t] = 0.f;
}

// --- 3. xl = x@W_l+b_l, xr = x@W_r+b_r (f16 out), WMMA ---------------------
__global__ __launch_bounds__(256) void proj_gemm(const float* __restrict__ x,
                                                 const _Float16* __restrict__ Wl16,
                                                 const _Float16* __restrict__ Wr16,
                                                 const float* __restrict__ bl,
                                                 const float* __restrict__ br,
                                                 _Float16* __restrict__ xl16,
                                                 _Float16* __restrict__ xr16, int N) {
    __shared__ __align__(32) _Float16 xs[16 * 128];  // A-fragment order
    int row0 = blockIdx.x * 16;
    bool full = (row0 + 16 <= N);  // block-uniform
    for (int i = threadIdx.x; i < 16 * 128; i += 256) {
        int r = i >> 7, c = i & 127;
        float v = (full || row0 + r < N) ? x[(size_t)(row0 + r) * 128 + c] : 0.f;
        xs[aswz_pos(r, c)] = (_Float16)v;
    }
    __syncthreads();
    int lane = threadIdx.x & 31, w = threadIdx.x >> 5;
    v16h a[4];
#pragma unroll
    for (int kc = 0; kc < 4; ++kc) a[kc] = ((const v16h*)xs)[(kc << 5) + lane];
    int n15 = lane & 15, kh = lane >> 4;
#pragma unroll 1
    for (int half = 0; half < 2; ++half) {  // 0: W_l -> xl16, 1: W_r -> xr16
        const _Float16* W = half ? Wr16 : Wl16;
        const float* bias = half ? br : bl;
        _Float16* dst     = half ? xr16 : xl16;
#pragma unroll 1
        for (int nt = w; nt < 32; nt += 8) {
            v16h bfr[4];
#pragma unroll
            for (int kc = 0; kc < 4; ++kc) bfr[kc] = load_b_swz(W, 4, nt, kc, lane);
            v8f c = {};
#pragma unroll
            for (int kc = 0; kc < 4; ++kc) c = wmma_f16(a[kc], bfr[kc], c);
            int n = nt * 16 + n15;
            float bv = bias[n];
            _Float16* dp = dst + (size_t)(row0 + kh * 8) * 512 + n;
            if (full) {
#pragma unroll
                for (int v = 0; v < 8; ++v) dp[(size_t)v * 512] = (_Float16)(c[v] + bv);
            } else {
#pragma unroll
                for (int v = 0; v < 8; ++v)
                    if (row0 + kh * 8 + v < N) dp[(size_t)v * 512] = (_Float16)(c[v] + bv);
            }
        }
    }
}

// --- 4. per-edge GATv2 score + segment max (one wave32 per edge) -----------
__global__ __launch_bounds__(256) void edge_score(const int* __restrict__ ei,
                                                  const _Float16* __restrict__ xl16,
                                                  const _Float16* __restrict__ xr16,
                                                  const float* __restrict__ att,
                                                  float* __restrict__ alpha,
                                                  unsigned* __restrict__ amaxu, int E, int N) {
    int e = blockIdx.x * 8 + (threadIdx.x >> 5);
    if (e >= E + N) return;
    int lane = threadIdx.x & 31;
    int src, dst;
    if (e < E) { src = ei[e]; dst = ei[E + e]; } else { src = dst = e - E; }
    const _Float16* pl = xl16 + (size_t)src * 512 + lane * 16;
    const _Float16* pr = xr16 + (size_t)dst * 512 + lane * 16;
    const float*    pa = att + lane * 16;  // head = lane>>3 (16 elems/lane, 128/head)
    float s = 0.f;
#pragma unroll
    for (int j = 0; j < 16; ++j) s += lrelu((float)pl[j] + (float)pr[j]) * pa[j];
    s += __shfl_xor(s, 1, 32);
    s += __shfl_xor(s, 2, 32);
    s += __shfl_xor(s, 4, 32);
    if ((lane & 7) == 0) {
        int h = lane >> 3;
        alpha[(size_t)e * 4 + h] = s;
        atomicMax(&amaxu[dst * 4 + h], fkey(s));
    }
}

// --- 5. exp(alpha - max) + segment sum -------------------------------------
__global__ void edge_exp(const int* __restrict__ ei, float* __restrict__ alpha,
                         const unsigned* __restrict__ amaxu, float* __restrict__ denom,
                         int E, int N) {
    int t = blockIdx.x * blockDim.x + threadIdx.x;
    if (t >= (E + N) * 4) return;
    int h = t & 3, e = t >> 2;
    int dst = (e < E) ? ei[E + e] : (e - E);
    float m  = funkey(amaxu[dst * 4 + h]);
    float ex = __expf(alpha[t] - m);
    alpha[t] = ex;
    atomicAdd(&denom[dst * 4 + h], ex);
}

// --- 6. message aggregation (head-mean folded in), one wave per edge -------
__global__ __launch_bounds__(256) void edge_aggregate(const int* __restrict__ ei,
                                                      const _Float16* __restrict__ xl16,
                                                      const float* __restrict__ alpha,
                                                      const float* __restrict__ denom,
                                                      float* __restrict__ acc, int E, int N) {
    int e = blockIdx.x * 8 + (threadIdx.x >> 5);
    if (e >= E + N) return;
    int lane = threadIdx.x & 31;
    int src, dst;
    if (e < E) { src = ei[e]; dst = ei[E + e]; } else { src = dst = e - E; }
    float w[4];
#pragma unroll
    for (int h = 0; h < 4; ++h)
        w[h] = 0.25f * alpha[(size_t)e * 4 + h] / (denom[dst * 4 + h] + 1e-16f);
#pragma unroll
    for (int k = 0; k < 4; ++k) {
        int c = k * 32 + lane;  // coalesced atomics across the wave
        float v = 0.f;
#pragma unroll
        for (int h = 0; h < 4; ++h) v += w[h] * (float)xl16[(size_t)src * 512 + h * 128 + c];
        atomicAdd(&acc[(size_t)dst * 128 + c], v);
    }
}

// --- 7. h = x + gat_out + gat_bias; LayerNorm -> hn (f32 + f16) ------------
__global__ __launch_bounds__(256) void ln_kernel(const float* __restrict__ x,
                                                 const float* __restrict__ acc,
                                                 const float* __restrict__ gbias,
                                                 const float* __restrict__ g,
                                                 const float* __restrict__ b,
                                                 float* __restrict__ hnf,
                                                 _Float16* __restrict__ hnh, int N) {
    int n = blockIdx.x * 8 + (threadIdx.x >> 5);
    if (n >= N) return;
    int lane = threadIdx.x & 31;
    float h[4], s = 0.f;
#pragma unroll
    for (int k = 0; k < 4; ++k) {
        int c = k * 32 + lane;
        h[k] = x[(size_t)n * 128 + c] + acc[(size_t)n * 128 + c] + gbias[c];
        s += h[k];
    }
#pragma unroll
    for (int o = 1; o < 32; o <<= 1) s += __shfl_xor(s, o, 32);
    float mu = s * (1.f / 128.f), vs = 0.f;
#pragma unroll
    for (int k = 0; k < 4; ++k) { float d = h[k] - mu; vs += d * d; }
#pragma unroll
    for (int o = 1; o < 32; o <<= 1) vs += __shfl_xor(vs, o, 32);
    float rs = rsqrtf(vs * (1.f / 128.f) + 1e-5f);
#pragma unroll
    for (int k = 0; k < 4; ++k) {
        int c = k * 32 + lane;
        float hv = (h[k] - mu) * rs * g[c] + b[c];
        hnf[(size_t)n * 128 + c] = hv;
        hnh[(size_t)n * 128 + c] = (_Float16)hv;
    }
}

// --- 8. fused FFN: out = hn + lrelu(hn@W1+b1)@W2 + b2 (WMMA x2) ------------
__global__ __launch_bounds__(256) void ffn_gemm(const _Float16* __restrict__ hnh,
                                                const float* __restrict__ hnf,
                                                const _Float16* __restrict__ W1h,
                                                const float* __restrict__ b1,
                                                const _Float16* __restrict__ W2h,
                                                const float* __restrict__ b2,
                                                float* __restrict__ out, int N) {
    __shared__ __align__(32) _Float16 at[16 * 128];  // hn tile, A-fragment order (K=128)
    __shared__ __align__(32) _Float16 t16[16 * 256]; // t tile,  A-fragment order (K=256)
    int row0 = blockIdx.x * 16;
    bool full = (row0 + 16 <= N);  // block-uniform
    for (int i = threadIdx.x; i < 16 * 128; i += 256) {
        int r = i >> 7, c = i & 127;
        _Float16 v = (full || row0 + r < N) ? hnh[(size_t)(row0 + r) * 128 + c] : (_Float16)0.f;
        at[aswz_pos(r, c)] = v;
    }
    __syncthreads();
    int lane = threadIdx.x & 31, w = threadIdx.x >> 5;
    int n15 = lane & 15, kh = lane >> 4;
    v16h a[4];
#pragma unroll
    for (int kc = 0; kc < 4; ++kc) a[kc] = ((const v16h*)at)[(kc << 5) + lane];
    // GEMM1: t = lrelu(hn @ W1 + b1)  (16 x 256), 2 column-tiles per wave;
    // result written into LDS directly in second-GEMM A-fragment order.
#pragma unroll
    for (int ti = 0; ti < 2; ++ti) {
        int nt = w + ti * 8;
        v16h bfr[4];
#pragma unroll
        for (int kc = 0; kc < 4; ++kc) bfr[kc] = load_b_swz(W1h, 4, nt, kc, lane);
        v8f c = {};
#pragma unroll
        for (int kc = 0; kc < 4; ++kc) c = wmma_f16(a[kc], bfr[kc], c);
        int n = nt * 16 + n15;  // column of t == K index of GEMM2
        float bv = b1[n];
#pragma unroll
        for (int v = 0; v < 8; ++v) {
            int m = kh * 8 + v;
            t16[aswz_pos(m, n)] = (_Float16)lrelu(c[v] + bv);
        }
    }
    __syncthreads();
    // GEMM2: out = hn + t @ W2 + b2  (16 x 128), 1 column-tile per wave, K=256
    {
        int nt = w;
        v16h bfr[8];
#pragma unroll
        for (int kc = 0; kc < 8; ++kc) bfr[kc] = load_b_swz(W2h, 8, nt, kc, lane);
        v8f c = {};
#pragma unroll
        for (int kc = 0; kc < 8; ++kc) {
            v16h af = ((const v16h*)t16)[(kc << 5) + lane];
            c = wmma_f16(af, bfr[kc], c);
        }
        int n = nt * 16 + n15;
        float bv = b2[n];
        size_t base = (size_t)(row0 + kh * 8) * 128 + n;
        if (full) {
#pragma unroll
            for (int v = 0; v < 8; ++v)
                out[base + (size_t)v * 128] = hnf[base + (size_t)v * 128] + c[v] + bv;
        } else {
#pragma unroll
            for (int v = 0; v < 8; ++v)
                if (row0 + kh * 8 + v < N)
                    out[base + (size_t)v * 128] = hnf[base + (size_t)v * 128] + c[v] + bv;
        }
    }
}

// ---------------------------------------------------------------------------
extern "C" void kernel_launch(void* const* d_in, const int* in_sizes, int n_in,
                              void* d_out, int out_size, void* d_ws, size_t ws_size,
                              hipStream_t stream) {
    const float* x   = (const float*)d_in[0];
    const int*   ei  = (const int*)d_in[1];
    const float* Wl  = (const float*)d_in[2];
    const float* bl  = (const float*)d_in[3];
    const float* Wr  = (const float*)d_in[4];
    const float* br  = (const float*)d_in[5];
    const float* att = (const float*)d_in[6];
    const float* gb  = (const float*)d_in[7];
    const float* lng = (const float*)d_in[8];
    const float* lnb = (const float*)d_in[9];
    const float* W1  = (const float*)d_in[10];
    const float* b1  = (const float*)d_in[11];
    const float* W2  = (const float*)d_in[12];
    const float* b2  = (const float*)d_in[13];
    float* out = (float*)d_out;
    (void)n_in; (void)out_size; (void)ws_size;

    int N = in_sizes[0] / 128;
    int E = in_sizes[1] / 2;
    int Etot = E + N;

    char* base = (char*)d_ws;
    size_t off = 0;
    auto seg = [&](size_t bytes) -> char* {
        off = (off + 255) & ~(size_t)255;
        char* p = base + off;
        off += bytes;
        return p;
    };
    _Float16* xl16  = (_Float16*)seg((size_t)N * 512 * 2);
    _Float16* xr16  = (_Float16*)seg((size_t)N * 512 * 2);
    _Float16* Wl16  = (_Float16*)seg((size_t)128 * 512 * 2);
    _Float16* Wr16  = (_Float16*)seg((size_t)128 * 512 * 2);
    _Float16* W1h   = (_Float16*)seg((size_t)128 * 256 * 2);
    _Float16* W2h   = (_Float16*)seg((size_t)256 * 128 * 2);
    float*    alpha = (float*)seg((size_t)Etot * 4 * 4);
    unsigned* amaxu = (unsigned*)seg((size_t)N * 4 * 4);
    float*    denom = (float*)seg((size_t)N * 4 * 4);
    float*    acc   = (float*)seg((size_t)N * 128 * 4);
    float*    hnf   = (float*)seg((size_t)N * 128 * 4);
    _Float16* hnh   = (_Float16*)seg((size_t)N * 128 * 2);

    int mtiles = (N + 15) / 16;

    prep_weights<<<dim3((128 * 512 + 255) / 256), dim3(256), 0, stream>>>(
        Wl, Wr, W1, W2, Wl16, Wr16, W1h, W2h);
    init_ws<<<dim3((unsigned)((N * 128 + 255) / 256)), dim3(256), 0, stream>>>(amaxu, denom, acc, N);
    proj_gemm<<<dim3((unsigned)mtiles), dim3(256), 0, stream>>>(
        x, Wl16, Wr16, bl, br, xl16, xr16, N);
    edge_score<<<dim3((unsigned)((Etot + 7) / 8)), dim3(256), 0, stream>>>(
        ei, xl16, xr16, att, alpha, amaxu, E, N);
    edge_exp<<<dim3((unsigned)((Etot * 4 + 255) / 256)), dim3(256), 0, stream>>>(
        ei, alpha, amaxu, denom, E, N);
    edge_aggregate<<<dim3((unsigned)((Etot + 7) / 8)), dim3(256), 0, stream>>>(
        ei, xl16, alpha, denom, acc, E, N);
    ln_kernel<<<dim3((unsigned)((N + 7) / 8)), dim3(256), 0, stream>>>(
        x, acc, gb, lng, lnb, hnf, hnh, N);
    ffn_gemm<<<dim3((unsigned)mtiles), dim3(256), 0, stream>>>(
        hnh, hnf, W1h, b1, W2h, b2, out, N);
}
